// Gate_90950227460274
// MI455X (gfx1250) — compile-verified, matrix operand order checked by reading, not run on hardware
//
#include <hip/hip_runtime.h>
#include <hip/hip_bf16.h>

// ---------------------------------------------------------------------------
// DeepSeek-V3 style gate for MI455X (gfx1250, wave32, WMMA).
//   logits = x[8192,7168] @ W[256,7168]^T   (bf16 WMMA, fp32 accumulate)
//   then group-limited top-k routing.
// B-chunk staging uses GLOBAL_LOAD_ASYNC_TO_LDS_B128 (ASYNCcnt) when the
// toolchain exposes the builtin; falls back to VGPR round-trip otherwise.
// ---------------------------------------------------------------------------

typedef __attribute__((ext_vector_type(16))) __bf16 v16bf;
typedef __attribute__((ext_vector_type(8)))  float  v8f;

// async-to-LDS builtin operand types: 16B int vector in global / LDS addrspace
typedef int v4i __attribute__((vector_size(16)));
typedef __attribute__((address_space(1))) v4i gas_v4i;   // global (__device__)
typedef __attribute__((address_space(3))) v4i las_v4i;   // LDS (__shared__)

constexpr int T_TOK = 8192;   // tokens
constexpr int DIM   = 7168;   // hidden dim
constexpr int NEXP  = 256;    // experts
constexpr int KC    = 64;     // K chunk per LDS stage (2 WMMA k-steps)
constexpr int ROWS  = 32;     // token rows per workgroup
constexpr int LDB   = 80;     // padded LDS row stride in bf16 elems (160B, 32B-aligned)

#if __has_builtin(__builtin_amdgcn_global_load_async_to_lds_b128) && \
    __has_builtin(__builtin_amdgcn_s_wait_asynccnt)
#define USE_ASYNC_LDS 1
#else
#define USE_ASYNC_LDS 0
#endif

// ---------------------------------------------------------------------------
// Kernel 1: convert router weight fp32 [256,7168] -> bf16 (row-major per expert)
// ---------------------------------------------------------------------------
__global__ void __launch_bounds__(256)
cvt_weight_kernel(const float* __restrict__ w, __bf16* __restrict__ wb) {
  const int idx = (blockIdx.x * 256 + threadIdx.x) * 8;
  const float4* s = (const float4*)(w + idx);
  float4 f0 = s[0];
  float4 f1 = s[1];
  union { __bf16 h[8]; uint4 u; } pk;
  pk.h[0] = (__bf16)f0.x; pk.h[1] = (__bf16)f0.y;
  pk.h[2] = (__bf16)f0.z; pk.h[3] = (__bf16)f0.w;
  pk.h[4] = (__bf16)f1.x; pk.h[5] = (__bf16)f1.y;
  pk.h[6] = (__bf16)f1.z; pk.h[7] = (__bf16)f1.w;
  *(uint4*)(wb + idx) = pk.u;
}

// ---------------------------------------------------------------------------
// Kernel 2: GEMM via v_wmma_f32_16x16x32_bf16.
// Workgroup: 256 threads = 8 waves arranged 2(M) x 4(N); tile 32 rows x 256 cols.
// Double-buffered LDS staging of A (x, converted fp32->bf16) and B (bf16 W).
// ---------------------------------------------------------------------------
__global__ void __launch_bounds__(256)
gate_gemm_kernel(const float* __restrict__ x,
                 const __bf16* __restrict__ wb,
                 float* __restrict__ logits) {
  __shared__ __align__(32) __bf16 ldsA[2][ROWS * LDB];   // ~10 KB
  __shared__ __align__(32) __bf16 ldsB[2][NEXP * LDB];   // ~80 KB

  const int tid  = threadIdx.x;
  const int lane = tid & 31;
  const int wv   = tid >> 5;
  const int mw   = wv & 1;        // M-wave: 0..1  (16 rows each)
  const int nw   = wv >> 1;       // N-wave: 0..3  (64 cols each)
  const int r0   = blockIdx.x * ROWS;

  const int m  = lane & 15;            // row/col within fragment
  const int kb = (lane >> 4) << 4;     // 0 or 16 (K half; contiguous k = 16h + s, same for A and B)

  v8f acc[4] = {};                     // 4 N-tiles of 16x16 fp32

  auto stage = [&](int buf, int k0) {
    // ---- A: 32 rows x 64 k of x (fp32) -> bf16 into LDS (needs convert) ----
    {
      const int row = tid >> 3;            // 0..31
      const int col = (tid & 7) << 3;      // 0,8,..,56
      const float4* src =
          (const float4*)(x + (size_t)(r0 + row) * DIM + k0 + col);
      float4 f0 = src[0];
      float4 f1 = src[1];
      union { __bf16 h[8]; uint4 u; } pk;
      pk.h[0] = (__bf16)f0.x; pk.h[1] = (__bf16)f0.y;
      pk.h[2] = (__bf16)f0.z; pk.h[3] = (__bf16)f0.w;
      pk.h[4] = (__bf16)f1.x; pk.h[5] = (__bf16)f1.y;
      pk.h[6] = (__bf16)f1.z; pk.h[7] = (__bf16)f1.w;
      *(uint4*)((char*)&ldsA[buf][0] + row * (LDB * 2) + col * 2) = pk.u;
    }
    // ---- B: 256 experts x 64 k of bf16 W into LDS (32 KB payload) ----
#pragma unroll
    for (int i = 0; i < 8; ++i) {
      const int c  = tid + i * 256;        // 0..2047 16B-chunks
      const int e  = c >> 3;               // expert row (8 chunks of 16B per row)
      const int cb = (c & 7) << 4;         // byte offset within row payload
      const char* gsrc = (const char*)wb + (size_t)e * (DIM * 2) +
                         (size_t)k0 * 2 + cb;
      char* ldst = (char*)&ldsB[buf][0] + e * (LDB * 2) + cb;
#if USE_ASYNC_LDS
      __builtin_amdgcn_global_load_async_to_lds_b128(
          (gas_v4i*)(void*)gsrc, (las_v4i*)(void*)ldst, 0, 0);
#else
      *(uint4*)ldst = *(const uint4*)gsrc;
#endif
    }
  };

  auto stage_fence = [&]() {
#if USE_ASYNC_LDS
    __builtin_amdgcn_s_wait_asynccnt(0);   // async B writes to LDS landed
#endif
    __syncthreads();
  };

  stage(0, 0);
  stage_fence();

  constexpr int NSTAGE = DIM / KC;       // 112
  for (int ks = 0; ks < NSTAGE; ++ks) {
    const int buf = ks & 1;
    if (ks + 1 < NSTAGE) stage(buf ^ 1, (ks + 1) * KC);  // prefetch next chunk

#pragma unroll
    for (int ksub = 0; ksub < 2; ++ksub) {
      // A fragment: lane = M row (mw*16+m), 16 consecutive bf16 at K half kb
      v16bf a = *(const v16bf*)((const char*)&ldsA[buf][0] +
                 (mw * 16 + m) * (LDB * 2) + (ksub * 32 + kb) * 2);
#pragma unroll
      for (int nt = 0; nt < 4; ++nt) {
        const int e = nw * 64 + nt * 16 + m;   // expert column for this lane
        v16bf b = *(const v16bf*)((const char*)&ldsB[buf][0] +
                   e * (LDB * 2) + (ksub * 32 + kb) * 2);
        acc[nt] = __builtin_amdgcn_wmma_f32_16x16x32_bf16(
            false, a, false, b, (short)0, acc[nt], false, false);
      }
    }
    stage_fence();  // buf^1 fully staged; everyone done reading buf
  }

  // C layout: lanes 0-15: VGPR j -> M=j; lanes 16-31: M=8+j; N = lane&15
  const int rbase = r0 + mw * 16 + ((lane >> 4) << 3);
#pragma unroll
  for (int nt = 0; nt < 4; ++nt) {
    const int col = nw * 64 + nt * 16 + m;
#pragma unroll
    for (int j = 0; j < 8; ++j) {
      logits[(size_t)(rbase + j) * NEXP + col] = acc[nt][j];
    }
  }
}

// ---------------------------------------------------------------------------
// Kernel 3: routing. One wave32 per token; lane owns 8 experts (group = 4 lanes).
// ---------------------------------------------------------------------------
__global__ void __launch_bounds__(256)
gate_route_kernel(const float* __restrict__ logits,
                  const float* __restrict__ bias,
                  float* __restrict__ out_w,
                  float* __restrict__ out_i) {
  const int lane = threadIdx.x & 31;
  const int wv   = threadIdx.x >> 5;
  const int t    = blockIdx.x * 8 + wv;

  const float4* lp = (const float4*)(logits + (size_t)t * NEXP + lane * 8);
  float4 l0 = lp[0], l1 = lp[1];
  const float4* bp = (const float4*)(bias + lane * 8);
  float4 b0 = bp[0], b1 = bp[1];

  float lv[8] = {l0.x, l0.y, l0.z, l0.w, l1.x, l1.y, l1.z, l1.w};
  float bv[8] = {b0.x, b0.y, b0.z, b0.w, b1.x, b1.y, b1.z, b1.w};
  float sc[8], sb[8];
#pragma unroll
  for (int i = 0; i < 8; ++i) {
    sc[i] = 1.0f / (1.0f + __expf(-lv[i]));   // sigmoid score (routing weight)
    sb[i] = sc[i] + bv[i];                    // biased score (selection only)
  }

  // ---- group score = sum of top-2 biased scores within group of 32 ----
  float m1 = -3.0e38f, m2 = -3.0e38f;
#pragma unroll
  for (int i = 0; i < 8; ++i) {
    float v = sb[i];
    if (v > m1) { m2 = m1; m1 = v; } else if (v > m2) { m2 = v; }
  }
#pragma unroll
  for (int off = 1; off <= 2; off <<= 1) {    // merge across 4 lanes of group
    float o1 = __shfl_xor(m1, off);
    float o2 = __shfl_xor(m2, off);
    if (o1 > m1) { m2 = fmaxf(m1, o2); m1 = o1; }
    else         { m2 = fmaxf(m2, o1); }
  }
  const float gs = m1 + m2;

  // ---- top-4 groups (tie-break: lower group index, like lax.top_k) ----
  const int g = lane >> 2;
  int rank = 0;
#pragma unroll
  for (int k = 0; k < 8; ++k) {
    float o = __shfl(gs, k * 4);
    rank += ((o > gs) || (o == gs && k < g)) ? 1 : 0;
  }
  if (rank >= 4) {
#pragma unroll
    for (int i = 0; i < 8; ++i) sb[i] = -1.0e30f;   // NEG_INF mask
  }

  // ---- top-8 experts over masked biased scores; gather unbiased scores ----
  float wout[8]; int iout[8]; float wsum = 0.0f;
#pragma unroll
  for (int j = 0; j < 8; ++j) {
    float bm = -3.0e38f; int bi = 0;
#pragma unroll
    for (int i = 0; i < 8; ++i) {
      if (sb[i] > bm) { bm = sb[i]; bi = i; }       // lowest index wins ties
    }
    int bidx = lane * 8 + bi;
#pragma unroll
    for (int off = 16; off >= 1; off >>= 1) {       // wave argmax
      float ov = __shfl_xor(bm, off);
      int   oi = __shfl_xor(bidx, off);
      if (ov > bm || (ov == bm && oi < bidx)) { bm = ov; bidx = oi; }
    }
    const int ii = bidx & 7;
    float cand = sc[0];
#pragma unroll
    for (int i = 1; i < 8; ++i) cand = (ii == i) ? sc[i] : cand;
    float wsel = __shfl(cand, bidx >> 3);           // unbiased score of winner
    if ((bidx >> 3) == lane) {                      // owner removes winner
#pragma unroll
      for (int i = 0; i < 8; ++i) if (ii == i) sb[i] = -2.0e30f;
    }
    wout[j] = wsel; iout[j] = bidx; wsum += wsel;
  }

  if (lane == 0) {
    const float s = 2.5f / wsum;                    // normalize + route_scale
#pragma unroll
    for (int j = 0; j < 8; ++j) {
      out_w[(size_t)t * 8 + j] = wout[j] * s;
      out_i[(size_t)t * 8 + j] = (float)iout[j];
    }
  }
}

// ---------------------------------------------------------------------------
// Launch: inputs are {start_pos, x, weight, bias}; output = weights(8192x8)
// then indices(8192x8), concatenated flat as float.
// ---------------------------------------------------------------------------
extern "C" void kernel_launch(void* const* d_in, const int* in_sizes, int n_in,
                              void* d_out, int out_size, void* d_ws, size_t ws_size,
                              hipStream_t stream) {
  const float* x    = (const float*)d_in[1];
  const float* wgt  = (const float*)d_in[2];
  const float* bias = (const float*)d_in[3];

  __bf16* wb      = (__bf16*)d_ws;                                  // 3.67 MB
  float*  logits  = (float*)((char*)d_ws + (size_t)NEXP * DIM * 2); // 8 MB

  float* out_w = (float*)d_out;
  float* out_i = out_w + (size_t)T_TOK * 8;

  cvt_weight_kernel<<<(NEXP * DIM) / (256 * 8), 256, 0, stream>>>(wgt, wb);
  gate_gemm_kernel<<<T_TOK / ROWS, 256, 0, stream>>>(x, wb, logits);
  gate_route_kernel<<<T_TOK / 8, 256, 0, stream>>>(logits, bias, out_w, out_i);
}
